// CrossAttention_55748675502311
// MI455X (gfx1250) — matrix operand order repaired and checked by reading, not compiled, not run on hardware
//
#include <hip/hip_runtime.h>
#include <hip/hip_bf16.h>

// CrossAttention (token-wise + feature-wise) for MI455X (gfx1250, wave32, WMMA).
// B=8 N=1024 C=768 H=12 Dh=64.  All matmuls on v_wmma_f32_16x16x32_bf16.
// LDS staging uses GLOBAL_LOAD_ASYNC_TO_LDS_B128 (ASYNCcnt) where the
// toolchain provides the builtin, with a synchronous b128 fallback.
// k_token double-buffers K/V tiles: tile t+1 streams into LDS while tile t
// computes, one barrier per tile.

#define BB 8
#define NN 1024
#define CC 768
#define HH 12
#define DHH 64
#define ATT_SCALE 0.125f

typedef __bf16 bf16;
typedef __attribute__((ext_vector_type(16))) __bf16 bf16x16;
typedef __attribute__((ext_vector_type(8)))  __bf16 bf16x8;
typedef __attribute__((ext_vector_type(4)))  __bf16 bf16x4;
typedef __attribute__((ext_vector_type(8)))  float  v8f;
typedef __attribute__((ext_vector_type(4)))  int    v4i;

#if defined(__has_builtin)
#if __has_builtin(__builtin_amdgcn_global_load_async_to_lds_b128) && \
    __has_builtin(__builtin_amdgcn_s_wait_asynccnt)
#define USE_ASYNC_LDS 1
#endif
#endif

#ifdef USE_ASYNC_LDS
typedef __attribute__((address_space(1))) v4i v4i_g;   // global 16B chunk
typedef __attribute__((address_space(3))) v4i v4i_l;   // LDS 16B chunk
#endif

// 16-byte global -> LDS copy; async (VGPR-bypassing, ASYNCcnt) when available.
__device__ __forceinline__ void copy16_to_lds(void* lds_dst, const void* gsrc) {
#ifdef USE_ASYNC_LDS
  __builtin_amdgcn_global_load_async_to_lds_b128(
      (v4i_g*)gsrc, (v4i_l*)lds_dst, 0, 0);
#else
  *(bf16x8*)lds_dst = *(const bf16x8*)gsrc;
#endif
}
__device__ __forceinline__ void lds_copy_join() {
#ifdef USE_ASYNC_LDS
  __builtin_amdgcn_s_wait_asynccnt(0);
#endif
}

union BF16Frag { bf16x16 v; bf16x8 h[2]; };

// A-matrix 16x32 bf16 fragment (ISA 7.12.2): lane m=L%16, halfwords 0..7 ->
// K = (L/16)*8 + j, halfwords 8..15 -> K = 16 + (L/16)*8 + j.  Two b128 loads.
__device__ __forceinline__ bf16x16 load_a_frag(const bf16* tile, int ld, int k0, int lane) {
  const int m = lane & 15, g = lane >> 4;
  const bf16* p = tile + m * ld + k0 + g * 8;
  BF16Frag u;
  u.h[0] = *(const bf16x8*)(p);
  u.h[1] = *(const bf16x8*)(p + 16);
  return u.v;
}

// B-matrix 32x16 bf16 fragment from a TRANSPOSED tile bT[n][k]:
// lane n=L%16, halfwords 0..7 -> K = (L/16)*8 + j, halfwords 8..15 -> K = 16 + (L/16)*8 + j.
__device__ __forceinline__ bf16x16 load_bT_frag(const bf16* tileT, int ld, int k0, int lane) {
  const int n = lane & 15, g = lane >> 4;
  const bf16* p = tileT + n * ld + k0 + g * 8;
  BF16Frag u;
  u.h[0] = *(const bf16x8*)(p);
  u.h[1] = *(const bf16x8*)(p + 16);
  return u.v;
}

__device__ __forceinline__ v8f wmma_bf16(bf16x16 a, bf16x16 b, v8f c) {
  return __builtin_amdgcn_wmma_f32_16x16x32_bf16(false, a, false, b, (short)0, c, false, false);
}

// float4 -> 4 contiguous bf16 (one b64 LDS store, pk-cvt pairs)
__device__ __forceinline__ void cvt_store4(bf16* dst, float4 f) {
  bf16x4 o;
  o.x = (bf16)f.x; o.y = (bf16)f.y; o.z = (bf16)f.z; o.w = (bf16)f.w;
  *(bf16x4*)dst = o;
}

// ---------------------------------------------------------------------------
// One-time weight transpose+convert: WT[n][k] = (bf16)W[k][col0+n], 768x768.
// grid = (12, 12) k-tiles x n-tiles, block = 256.
__global__ __launch_bounds__(256) void k_wtr(const float* __restrict__ W, int ldw, int col0,
                                             bf16* __restrict__ WT) {
  const int tk = blockIdx.x, tn = blockIdx.y, tid = threadIdx.x;
#pragma unroll
  for (int j = 0; j < 4; ++j) {
    int idx = (tid + j * 256) * 4;          // 4096 elements per 64x64 tile
    int n = idx & 63, k = idx >> 6;
    float4 f = *(const float4*)&W[(size_t)(tk * 64 + k) * ldw + col0 + tn * 64 + n];
    WT[(size_t)(tn * 64 + n + 0) * CC + tk * 64 + k] = (bf16)f.x;
    WT[(size_t)(tn * 64 + n + 1) * CC + tk * 64 + k] = (bf16)f.y;
    WT[(size_t)(tn * 64 + n + 2) * CC + tk * 64 + k] = (bf16)f.z;
    WT[(size_t)(tn * 64 + n + 3) * CC + tk * 64 + k] = (bf16)f.w;
  }
}

// ---------------------------------------------------------------------------
// Projection GEMM: outN[b,h,n,d] = sum_k A[b*N+n,k] * WT[h*64+d][k]  (bf16).
// Optionally also writes transposed copy outT[b,h,d,n] (for V).
// grid = (C/64, B*N/64) with column tile == head; block = 128 (4 waves).
__global__ __launch_bounds__(128) void k_proj(const float* __restrict__ A,
                                              const bf16* __restrict__ WT,
                                              bf16* __restrict__ outN,
                                              bf16* __restrict__ outT) {
  __shared__ __align__(16) bf16 As[64][64];   // [m][k]
  __shared__ __align__(16) bf16 Bs[64][64];   // [n][k] (WT tile copy)
  const int tn = blockIdx.x, tm = blockIdx.y;
  const int tid = threadIdx.x, lane = tid & 31, w = tid >> 5;
  const int g = lane >> 4, nl = lane & 15;
  v8f acc[4] = {};
  for (int k0 = 0; k0 < CC; k0 += 64) {
    if (k0 + 64 < CC)   // prefetch next A tile (f32 path keeps VGPR staging)
      __builtin_prefetch(&A[(size_t)(tm * 64 + (tid >> 1)) * CC + k0 + 64 + (tid & 1) * 32], 0, 3);
    __syncthreads();
#pragma unroll
    for (int j = 0; j < 4; ++j) {            // WT tile: async memory->LDS copy
      int vi = tid + j * 128;
      int n = vi >> 3, c8 = (vi & 7) * 8;
      copy16_to_lds(&Bs[n][c8], &WT[(size_t)(tn * 64 + n) * CC + k0 + c8]);
    }
    // A: issue all 8 b128 loads first (clause, one wait), then cvt+store
    float4 fa[8];
#pragma unroll
    for (int j = 0; j < 8; ++j) {
      int idx = (tid + j * 128) * 4;
      int r = idx >> 6, c = idx & 63;
      fa[j] = *(const float4*)&A[(size_t)(tm * 64 + r) * CC + k0 + c];
    }
#pragma unroll
    for (int j = 0; j < 8; ++j) {
      int idx = (tid + j * 128) * 4;
      int r = idx >> 6, c = idx & 63;
      cvt_store4(&As[r][c], fa[j]);
    }
    lds_copy_join();
    __syncthreads();
    bf16x16 af0 = load_a_frag(&As[w * 16][0], 64, 0, lane);
    bf16x16 af1 = load_a_frag(&As[w * 16][0], 64, 32, lane);
    bf16x16 bfr[8];
#pragma unroll
    for (int kk = 0; kk < 2; ++kk)
#pragma unroll
      for (int nn = 0; nn < 4; ++nn)
        bfr[kk * 4 + nn] = load_bT_frag(&Bs[nn * 16][0], 64, kk * 32, lane);
#pragma unroll
    for (int nn = 0; nn < 4; ++nn) {
      acc[nn] = wmma_bf16(af0, bfr[nn], acc[nn]);
      acc[nn] = wmma_bf16(af1, bfr[4 + nn], acc[nn]);
    }
  }
  // writeback: restage through LDS so global stores are coalesced b128
  __syncthreads();
  const int bO = (tm * 64) >> 10;             // batch constant per block
  const int nbase = (tm * 64) & (NN - 1);
#pragma unroll
  for (int nn = 0; nn < 4; ++nn)
#pragma unroll
    for (int r = 0; r < 8; ++r) {
      bf16 v = (bf16)acc[nn][r];
      int rl = w * 16 + g * 8 + r, cl = nn * 16 + nl;
      As[rl][cl] = v;                         // natural [n][d]
      if (outT) Bs[cl][rl] = v;               // transposed [d][n]
    }
  __syncthreads();
  bf16* dstN = outN + (((size_t)bO * HH + tn) * NN + nbase) * DHH;
#pragma unroll
  for (int j = 0; j < 4; ++j) {
    int vi = tid + j * 128;
    int rl = vi >> 3, c8 = (vi & 7) * 8;
    *(bf16x8*)&dstN[(size_t)rl * DHH + c8] = *(const bf16x8*)&As[rl][c8];
  }
  if (outT) {
    bf16* dstT = outT + (((size_t)bO * HH + tn) * DHH) * NN + nbase;
#pragma unroll
    for (int j = 0; j < 4; ++j) {
      int vi = tid + j * 128;
      int dl = vi >> 3, n8 = (vi & 7) * 8;
      *(bf16x8*)&dstT[(size_t)dl * NN + n8] = *(const bf16x8*)&Bs[dl][n8];
    }
  }
}

// ---------------------------------------------------------------------------
// Feature-wise attention: G = Q^T K (64x64, K-dim N), softmax rows,
// f = V @ P^T -> TF[b,n, h*64+d]  (writes TF).  grid = (H, B), block = 128.
__global__ __launch_bounds__(128) void k_feat(const bf16* __restrict__ Q,
                                              const bf16* __restrict__ K,
                                              const bf16* __restrict__ V,
                                              float* __restrict__ TF) {
  __shared__ __align__(16) bf16 Qt[64][64];   // [d][n_local]
  __shared__ __align__(16) bf16 Kt[64][64];   // [e][n_local]
  __shared__ __align__(16) bf16 Ps[64][64];   // softmax(G)[d][e]
  const int h = blockIdx.x, b = blockIdx.y;
  const int tid = threadIdx.x, lane = tid & 31, w = tid >> 5;
  const int g = lane >> 4, nl = lane & 15;
  const size_t head = ((size_t)b * HH + h) * NN * DHH;
  const bf16* Qh = Q + head;
  const bf16* Kh = K + head;
  const bf16* Vh = V + head;

  // Phase 1: G[d,e] = sum_n Q[n,d] K[n,e], n-step 64
  v8f acc[4] = {};
  for (int n0 = 0; n0 < NN; n0 += 64) {
    if (n0 + 64 < NN) {
      __builtin_prefetch(&Qh[(size_t)(n0 + 64) * DHH + tid * 32], 0, 3);
      __builtin_prefetch(&Kh[(size_t)(n0 + 64) * DHH + tid * 32], 0, 3);
    }
    __syncthreads();
#pragma unroll
    for (int j = 0; j < 4; ++j) {            // b128 loads along d, transposed stores
      int vi = tid + j * 128;                // 512 vectors of 8
      int n = vi >> 3, d8 = (vi & 7) * 8;
      bf16x8 q8 = *(const bf16x8*)&Qh[(size_t)(n0 + n) * DHH + d8];
      bf16x8 k8 = *(const bf16x8*)&Kh[(size_t)(n0 + n) * DHH + d8];
#pragma unroll
      for (int t = 0; t < 8; ++t) { Qt[d8 + t][n] = q8[t]; Kt[d8 + t][n] = k8[t]; }
    }
    __syncthreads();
    bf16x16 af0 = load_a_frag(&Qt[w * 16][0], 64, 0, lane);
    bf16x16 af1 = load_a_frag(&Qt[w * 16][0], 64, 32, lane);
    bf16x16 bfr[8];
#pragma unroll
    for (int kk = 0; kk < 2; ++kk)
#pragma unroll
      for (int ee = 0; ee < 4; ++ee)
        bfr[kk * 4 + ee] = load_bT_frag(&Kt[ee * 16][0], 64, kk * 32, lane);
#pragma unroll
    for (int ee = 0; ee < 4; ++ee) {
      acc[ee] = wmma_bf16(af0, bfr[ee], acc[ee]);
      acc[ee] = wmma_bf16(af1, bfr[4 + ee], acc[ee]);
    }
  }
  // Phase 2: softmax over e (rows d), normalized P -> LDS bf16
#pragma unroll
  for (int r = 0; r < 8; ++r) {
    float mx = -1e30f;
#pragma unroll
    for (int ee = 0; ee < 4; ++ee) { acc[ee][r] *= ATT_SCALE; mx = fmaxf(mx, acc[ee][r]); }
#pragma unroll
    for (int off = 1; off < 16; off <<= 1) mx = fmaxf(mx, __shfl_xor(mx, off));
    float rs = 0.f;
#pragma unroll
    for (int ee = 0; ee < 4; ++ee) { float p = __expf(acc[ee][r] - mx); acc[ee][r] = p; rs += p; }
#pragma unroll
    for (int off = 1; off < 16; off <<= 1) rs += __shfl_xor(rs, off);
    float inv = 1.f / rs;
#pragma unroll
    for (int ee = 0; ee < 4; ++ee)
      Ps[w * 16 + g * 8 + r][ee * 16 + nl] = (bf16)(acc[ee][r] * inv);
  }
  __syncthreads();
  // Phase 3: f[n,d] = sum_e V[n,e] P[d,e]  (B-frags hoisted; A straight from global)
  bf16x16 pb[2][4];
#pragma unroll
  for (int kk = 0; kk < 2; ++kk)
#pragma unroll
    for (int dd = 0; dd < 4; ++dd)
      pb[kk][dd] = load_bT_frag(&Ps[dd * 16][0], 64, kk * 32, lane);
  for (int nb = 0; nb < NN / 64; ++nb) {
    const bf16* Ar = Vh + (size_t)(nb * 64 + w * 16) * DHH;
    bf16x16 a0 = load_a_frag(Ar, 64, 0, lane);
    bf16x16 a1 = load_a_frag(Ar, 64, 32, lane);
    v8f f[4] = {};
#pragma unroll
    for (int dd = 0; dd < 4; ++dd) {
      f[dd] = wmma_bf16(a0, pb[0][dd], f[dd]);
      f[dd] = wmma_bf16(a1, pb[1][dd], f[dd]);
    }
#pragma unroll
    for (int dd = 0; dd < 4; ++dd)
#pragma unroll
      for (int r = 0; r < 8; ++r) {
        int n = nb * 64 + w * 16 + g * 8 + r;
        int col = h * 64 + dd * 16 + nl;
        TF[((size_t)b * NN + n) * CC + col] = f[dd][r];
      }
  }
}

// ---------------------------------------------------------------------------
// Token-wise flash attention, double-buffered: K/V tiles for step t+1 stream
// into LDS (async) while step t computes.  V pre-transposed [b,h,d,n].
// Accumulates into TF (feature part already written).  grid = (N/64, H, B).
__global__ __launch_bounds__(128) void k_token(const bf16* __restrict__ Q,
                                               const bf16* __restrict__ K,
                                               const bf16* __restrict__ VT,
                                               float* __restrict__ TF) {
  __shared__ __align__(16) bf16 Qs[64][64];       // [m][d]
  __shared__ __align__(16) bf16 Ks[2][64][64];    // [key][d], double buffer
  __shared__ __align__(16) bf16 Vt[2][64][64];    // [d][key], double buffer
  __shared__ __align__(16) bf16 Ps[64][64];       // P operand restage
  const int qb = blockIdx.x, h = blockIdx.y, b = blockIdx.z;
  const int tid = threadIdx.x, lane = tid & 31, w = tid >> 5;
  const int g = lane >> 4, nl = lane & 15;
  const size_t head = ((size_t)b * HH + h) * NN * DHH;
  const bf16* Qh = Q + head + (size_t)qb * 64 * DHH;
  const bf16* Kh = K + head;
  const bf16* VTh = VT + head;                    // [d][n] layout, same extent

  for (int i = tid; i < 64 * 64 / 8; i += 128)    // Q block: contiguous b128 copy
    ((bf16x8*)&Qs[0][0])[i] = ((const bf16x8*)Qh)[i];
#pragma unroll
  for (int j = 0; j < 4; ++j) {                   // stage K/V tile 0 (async)
    int vi = tid + j * 128;
    copy16_to_lds((bf16*)&Ks[0][0][0] + vi * 8, Kh + vi * 8);
    int d = vi >> 3, c8 = (vi & 7) * 8;
    copy16_to_lds(&Vt[0][d][c8], &VTh[(size_t)d * NN + c8]);
  }
  lds_copy_join();
  __syncthreads();
  bf16x16 qa0 = load_a_frag(&Qs[w * 16][0], 64, 0, lane);
  bf16x16 qa1 = load_a_frag(&Qs[w * 16][0], 64, 32, lane);

  float m_i[8], l_i[8];
  v8f o[4] = {};
#pragma unroll
  for (int r = 0; r < 8; ++r) { m_i[r] = -1e30f; l_i[r] = 0.f; }

  for (int kt = 0; kt < NN / 64; ++kt) {
    const int cur = kt & 1;
    if (kt + 1 < NN / 64) {                       // stream next tile while computing
      const int nxt = cur ^ 1;
      const bf16* Kn = Kh + (size_t)(kt + 1) * 64 * DHH;
#pragma unroll
      for (int j = 0; j < 4; ++j) {
        int vi = tid + j * 128;
        copy16_to_lds((bf16*)&Ks[nxt][0][0] + vi * 8, Kn + vi * 8);
        int d = vi >> 3, c8 = (vi & 7) * 8;
        copy16_to_lds(&Vt[nxt][d][c8], &VTh[(size_t)d * NN + (kt + 1) * 64 + c8]);
      }
    }

    // S = Q @ K^T  (per wave: 16 rows x 64 keys; batch frags, then 8 WMMAs)
    bf16x16 kb[8];
#pragma unroll
    for (int kk = 0; kk < 2; ++kk)
#pragma unroll
      for (int nn = 0; nn < 4; ++nn)
        kb[kk * 4 + nn] = load_bT_frag(&Ks[cur][nn * 16][0], 64, kk * 32, lane);
    v8f s[4];
#pragma unroll
    for (int nn = 0; nn < 4; ++nn) {
      v8f z = {};
      z = wmma_bf16(qa0, kb[nn], z);
      z = wmma_bf16(qa1, kb[4 + nn], z);
      s[nn] = z;
    }
    // online softmax (rows live in 16-lane halves -> shfl_xor 1..8)
    float alpha[8];
#pragma unroll
    for (int r = 0; r < 8; ++r) {
      float mx = -1e30f;
#pragma unroll
      for (int nn = 0; nn < 4; ++nn) { s[nn][r] *= ATT_SCALE; mx = fmaxf(mx, s[nn][r]); }
#pragma unroll
      for (int off = 1; off < 16; off <<= 1) mx = fmaxf(mx, __shfl_xor(mx, off));
      float mnew = fmaxf(m_i[r], mx);
      alpha[r] = __expf(m_i[r] - mnew);
      m_i[r] = mnew;
      float rs = 0.f;
#pragma unroll
      for (int nn = 0; nn < 4; ++nn) { float p = __expf(s[nn][r] - mnew); s[nn][r] = p; rs += p; }
#pragma unroll
      for (int off = 1; off < 16; off <<= 1) rs += __shfl_xor(rs, off);
      l_i[r] = l_i[r] * alpha[r] + rs;
    }
    // restage P through LDS so it can re-enter WMMA as A operand (intra-wave dep)
#pragma unroll
    for (int nn = 0; nn < 4; ++nn)
#pragma unroll
      for (int r = 0; r < 8; ++r)
        Ps[w * 16 + g * 8 + r][nn * 16 + nl] = (bf16)s[nn][r];
#pragma unroll
    for (int dd = 0; dd < 4; ++dd)
#pragma unroll
      for (int r = 0; r < 8; ++r)
        o[dd][r] *= alpha[r];
    bf16x16 pa0 = load_a_frag(&Ps[w * 16][0], 64, 0, lane);
    bf16x16 pa1 = load_a_frag(&Ps[w * 16][0], 64, 32, lane);
    bf16x16 vb[8];
#pragma unroll
    for (int kk = 0; kk < 2; ++kk)
#pragma unroll
      for (int dd = 0; dd < 4; ++dd)
        vb[kk * 4 + dd] = load_bT_frag(&Vt[cur][dd * 16][0], 64, kk * 32, lane);
#pragma unroll
    for (int dd = 0; dd < 4; ++dd) {
      o[dd] = wmma_bf16(pa0, vb[dd], o[dd]);
      o[dd] = wmma_bf16(pa1, vb[4 + dd], o[dd]);
    }
    lds_copy_join();       // next tile resident before anyone reads it
    __syncthreads();
  }
  // normalize and accumulate on top of the feature part
#pragma unroll
  for (int dd = 0; dd < 4; ++dd)
#pragma unroll
    for (int r = 0; r < 8; ++r) {
      int n = qb * 64 + w * 16 + g * 8 + r;
      int col = h * 64 + dd * 16 + nl;
      size_t idx = ((size_t)b * NN + n) * CC + col;
      TF[idx] = TF[idx] + o[dd][r] / l_i[r];
    }
}

// ---------------------------------------------------------------------------
// Output projection: out = TF @ Wproj + bias (f32 out), WprojT pre-transposed.
// grid=(C/64, B*N/64).
__global__ __launch_bounds__(128) void k_out(const float* __restrict__ TF,
                                             const bf16* __restrict__ WT,
                                             const float* __restrict__ bias,
                                             float* __restrict__ out) {
  __shared__ __align__(16) bf16 As[64][64];
  __shared__ __align__(16) bf16 Bs[64][64];
  const int tn = blockIdx.x, tm = blockIdx.y;
  const int tid = threadIdx.x, lane = tid & 31, w = tid >> 5;
  const int g = lane >> 4, nl = lane & 15;
  v8f acc[4] = {};
  for (int k0 = 0; k0 < CC; k0 += 64) {
    if (k0 + 64 < CC)
      __builtin_prefetch(&TF[(size_t)(tm * 64 + (tid >> 1)) * CC + k0 + 64 + (tid & 1) * 32], 0, 3);
    __syncthreads();
#pragma unroll
    for (int j = 0; j < 4; ++j) {            // WT tile: async memory->LDS copy
      int vi = tid + j * 128;
      int n = vi >> 3, c8 = (vi & 7) * 8;
      copy16_to_lds(&Bs[n][c8], &WT[(size_t)(tn * 64 + n) * CC + k0 + c8]);
    }
    float4 fa[8];
#pragma unroll
    for (int j = 0; j < 8; ++j) {
      int idx = (tid + j * 128) * 4;
      int r = idx >> 6, c = idx & 63;
      fa[j] = *(const float4*)&TF[(size_t)(tm * 64 + r) * CC + k0 + c];
    }
#pragma unroll
    for (int j = 0; j < 8; ++j) {
      int idx = (tid + j * 128) * 4;
      int r = idx >> 6, c = idx & 63;
      cvt_store4(&As[r][c], fa[j]);
    }
    lds_copy_join();
    __syncthreads();
    bf16x16 af0 = load_a_frag(&As[w * 16][0], 64, 0, lane);
    bf16x16 af1 = load_a_frag(&As[w * 16][0], 64, 32, lane);
    bf16x16 bfr[8];
#pragma unroll
    for (int kk = 0; kk < 2; ++kk)
#pragma unroll
      for (int nn = 0; nn < 4; ++nn)
        bfr[kk * 4 + nn] = load_bT_frag(&Bs[nn * 16][0], 64, kk * 32, lane);
#pragma unroll
    for (int nn = 0; nn < 4; ++nn) {
      acc[nn] = wmma_bf16(af0, bfr[nn], acc[nn]);
      acc[nn] = wmma_bf16(af1, bfr[4 + nn], acc[nn]);
    }
  }
#pragma unroll
  for (int nn = 0; nn < 4; ++nn)
#pragma unroll
    for (int r = 0; r < 8; ++r) {
      int row = tm * 64 + w * 16 + g * 8 + r;
      int col = tn * 64 + nn * 16 + nl;
      out[(size_t)row * CC + col] = acc[nn][r] + bias[col];
    }
}

// ---------------------------------------------------------------------------
extern "C" void kernel_launch(void* const* d_in, const int* in_sizes, int n_in,
                              void* d_out, int out_size, void* d_ws, size_t ws_size,
                              hipStream_t stream) {
  const float* q     = (const float*)d_in[0];
  const float* k     = (const float*)d_in[1];
  const float* Wq    = (const float*)d_in[2];
  const float* Wkv   = (const float*)d_in[3];
  const float* Wproj = (const float*)d_in[4];
  const float* bproj = (const float*)d_in[5];
  float* out = (float*)d_out;

  char* ws = (char*)d_ws;
  const size_t QKV_BYTES = (size_t)BB * HH * NN * DHH * sizeof(bf16);  // 12.6 MB
  const size_t TF_BYTES  = (size_t)BB * NN * CC * sizeof(float);       // 25.2 MB
  const size_t WT_BYTES  = (size_t)CC * CC * sizeof(bf16);             // 1.18 MB
  bf16*  Qws  = (bf16*)(ws);
  bf16*  Kws  = (bf16*)(ws + QKV_BYTES);
  bf16*  Vws  = (bf16*)(ws + 2 * QKV_BYTES);
  bf16*  VTws = (bf16*)(ws + 3 * QKV_BYTES);
  float* TF   = (float*)(ws + 4 * QKV_BYTES);
  bf16*  WqT  = (bf16*)(ws + 4 * QKV_BYTES + TF_BYTES);
  bf16*  WkT  = (bf16*)(ws + 4 * QKV_BYTES + TF_BYTES + WT_BYTES);
  bf16*  WvT  = (bf16*)(ws + 4 * QKV_BYTES + TF_BYTES + 2 * WT_BYTES);
  bf16*  WpT  = (bf16*)(ws + 4 * QKV_BYTES + TF_BYTES + 3 * WT_BYTES);

  dim3 blk(128);
  dim3 gproj(CC / 64, (BB * NN) / 64);
  dim3 gwt(CC / 64, CC / 64);
  k_wtr<<<gwt, 256, 0, stream>>>(Wq,    CC,     0,  WqT);
  k_wtr<<<gwt, 256, 0, stream>>>(Wkv,   2 * CC, 0,  WkT);
  k_wtr<<<gwt, 256, 0, stream>>>(Wkv,   2 * CC, CC, WvT);
  k_wtr<<<gwt, 256, 0, stream>>>(Wproj, CC,     0,  WpT);
  k_proj<<<gproj, blk, 0, stream>>>(q, WqT, Qws, (bf16*)nullptr);
  k_proj<<<gproj, blk, 0, stream>>>(k, WkT, Kws, (bf16*)nullptr);
  k_proj<<<gproj, blk, 0, stream>>>(k, WvT, Vws, VTws);
  k_feat <<<dim3(HH, BB),          blk, 0, stream>>>(Qws, Kws, Vws, TF);
  k_token<<<dim3(NN / 64, HH, BB), blk, 0, stream>>>(Qws, Kws, VTws, TF);
  k_out  <<<gproj, blk, 0, stream>>>(TF, WpT, bproj, out);
}
